// RSNN_d_d_28028956573975
// MI455X (gfx1250) — matrix operand order repaired and checked by reading, not compiled, not run on hardware
//
#include <hip/hip_runtime.h>
#include <stdint.h>

// ---------------- problem constants ----------------
#define B_      256
#define WIN_    100
#define NIN_    700
#define H_      512
#define NOUT_   20
#define ND_     10
#define HD_     5120          // H*ND
#define MAXD_   40
#define THRESH_ 0.3f

#define KT1_    22            // ceil(704/32) k-tiles for GEMM1 (NIN padded to 704)
#define KT2_    160           // 5120/32 k-tiles for GEMM2/GEMM3
#define NT_H_   32            // 512/16 n-tiles
#define NT_O_   2             // 32/16 n-tiles (NOUT padded to 32)

#define NCHUNK_ 4             // prev-spike A buffer chunking
#define CKT_    40            // k-tiles per chunk (4*40 = 160)
#define CWORDS_ (CKT_ * 256)  // 10240 words = 40 KB per chunk buffer

// fragment-packed word counts in workspace
#define W0F_WORDS (KT1_ * NT_H_ * 256)        // 180224
#define W1F_WORDS (KT2_ * NT_H_ * 256)        // 1310720
#define WOF_WORDS (KT2_ * NT_O_ * 256)        // 81920
#define XF_WORDS  (WIN_ * 16 * KT1_ * 256)    // 9011200

typedef _Float16     v16h __attribute__((ext_vector_type(16)));
typedef float        v8f  __attribute__((ext_vector_type(8)));
typedef unsigned int v8u  __attribute__((ext_vector_type(8)));

struct Frag { uint4 lo, hi; };

__device__ __forceinline__ Frag ldfrag(const uint32_t* base, int lane) {
  const uint4* p = (const uint4*)base;
  Frag f;
  f.lo = p[lane];
  f.hi = p[32 + lane];
  return f;
}

__device__ __forceinline__ v16h frag16(uint4 lo, uint4 hi) {
  v8u u = {lo.x, lo.y, lo.z, lo.w, hi.x, hi.y, hi.z, hi.w};
  return __builtin_bit_cast(v16h, u);
}

// D = A(16x32 f16) * B(32x16 f16) + C(16x16 f32)
__device__ __forceinline__ v8f wmma_f(const Frag& a, const Frag& b, v8f c) {
  return __builtin_amdgcn_wmma_f32_16x16x32_f16(false, frag16(a.lo, a.hi),
                                                false, frag16(b.lo, b.hi),
                                                (short)0, c, false, false);
}

__device__ __forceinline__ uint32_t pack2h(float f0, float f1) {
  _Float16 h0 = (_Float16)f0;
  _Float16 h1 = (_Float16)f1;
  return (uint32_t)__builtin_bit_cast(unsigned short, h0) |
         ((uint32_t)__builtin_bit_cast(unsigned short, h1) << 16);
}

// ------------- pre-pass: pack W (B-matrix fragment layout, K-striped) -------------
__global__ void pack_b_f16(const float* __restrict__ src, uint32_t* __restrict__ dst,
                           int Nrows, int K, int KT, int NT) {
  int idx = blockIdx.x * blockDim.x + threadIdx.x;
  int total = KT * NT * 256;
  if (idx >= total) return;
  int tile = idx >> 8;                 // kt*NT + nt
  int rem  = idx & 255;                // c*128 + lane*4 + q
  int lane = (rem >> 2) & 31;
  int q    = rem & 3;
  int c    = rem >> 7;
  int v    = (c << 2) | q;             // vgpr index 0..7
  int kt = tile / NT, nt = tile - kt * NT;
  int n  = nt * 16 + (lane & 15);
  int jB = (v & 3) + ((lane & 16) ? 4 : 0) + ((v & 4) ? 8 : 0);
  int k0 = kt * 32 + 2 * jB;
  float f0 = (n < Nrows && k0     < K) ? src[(size_t)n * K + k0    ] : 0.0f;
  float f1 = (n < Nrows && k0 + 1 < K) ? src[(size_t)n * K + k0 + 1] : 0.0f;
  dst[idx] = pack2h(f0, f1);
}

// ------------- pre-pass: pack input x (A-matrix fragment layout) -------------
__global__ void pack_x_f16(const float* __restrict__ input, uint32_t* __restrict__ dst) {
  int idx = blockIdx.x * blockDim.x + threadIdx.x;
  if (idx >= XF_WORDS) return;
  int rem  = idx & 255;
  int tile = idx >> 8;                 // (t*16 + bt)*KT1 + kt
  int kt = tile % KT1_;
  int tb = tile / KT1_;
  int bt = tb & 15;
  int t  = tb >> 4;
  int lane = (rem >> 2) & 31;
  int q = rem & 3, c = rem >> 7;
  int v = (c << 2) | q;
  int m = lane & 15;
  int b = bt * 16 + m;
  int jA = (v & 3) + ((lane & 16) ? 4 : 0) + ((v & 4) ? 8 : 0);
  int k0 = kt * 32 + 2 * jA;
  float f0 = (k0     < NIN_) ? input[((size_t)b * WIN_ + t) * NIN_ + k0    ] : 0.0f;
  float f1 = (k0 + 1 < NIN_) ? input[((size_t)b * WIN_ + t) * NIN_ + k0 + 1] : 0.0f;
  dst[idx] = pack2h(f0, f1);
}

// ------------- spike-bit expansion (one 40-k-tile chunk) -------------
// prev[b, k] with k = h*10 + d comes from bitbuf physical slice (tslot + 4d) % 40, row b, bit h.
__device__ __forceinline__ void expand_chunk(const uint32_t* bb, uint32_t* ab,
                                             int tslot, int tid, int chunk) {
  const int rem  = tid & 255;
  const int lane = (rem >> 2) & 31;
  const int q = rem & 3, c = rem >> 7;
  const int v = (c << 2) | q;
  const int m = lane & 15;
  const int jA = (v & 3) + ((lane & 16) ? 4 : 0) + ((v & 4) ? 8 : 0);
  int k0 = (chunk * CKT_ + (tid >> 8)) * 32 + 2 * jA;
  int h = k0 / 10;
  int d = k0 - 10 * h;
  const int mbase = m * 16;
  uint32_t* abp = ab + tid;
  #pragma unroll
  for (int i = 0; i < CWORDS_ / 1024; ++i) {   // 10 words per thread
    const int carry = (d == 9) ? 1 : 0;
    const int h1 = h + carry;
    const int d1 = carry ? 0 : d + 1;
    int sl0 = tslot + 4 * d;  if (sl0 >= MAXD_) sl0 -= MAXD_;
    int sl1 = tslot + 4 * d1; if (sl1 >= MAXD_) sl1 -= MAXD_;
    uint32_t w0 = bb[sl0 * 256 + mbase + (h  >> 5)];
    uint32_t w1 = bb[sl1 * 256 + mbase + (h1 >> 5)];
    uint32_t b0 = (w0 >> (h  & 31)) & 1u;
    uint32_t b1 = (w1 >> (h1 & 31)) & 1u;
    *abp = (b0 ? 0x3C00u : 0u) | (b1 ? 0x3C000000u : 0u);
    abp += 1024;
    h += 12; d += 8;                           // k += 128
    if (d >= 10) { d -= 10; h += 1; }
  }
}

// Ping-pong pipelined GEMM over KT k-tiles (KT even, >= 4).
template <bool A_LDS>
__device__ __forceinline__ v8f gemm_pipe(const uint32_t* __restrict__ abase,
                                         const uint32_t* __restrict__ bbase,
                                         int bstride_words, int KT, int lane, v8f acc) {
  Frag a0 = ldfrag(abase, lane);
  Frag b0 = ldfrag(bbase, lane);
  Frag a1 = ldfrag(abase + 256, lane);
  Frag b1 = ldfrag(bbase + (size_t)bstride_words, lane);
  #pragma unroll 1
  for (int kt = 0; kt < KT - 2; kt += 2) {
    acc = wmma_f(a0, b0, acc);
    a0 = ldfrag(abase + (size_t)(kt + 2) * 256, lane);
    b0 = ldfrag(bbase + (size_t)(kt + 2) * bstride_words, lane);
    acc = wmma_f(a1, b1, acc);
    a1 = ldfrag(abase + (size_t)(kt + 3) * 256, lane);
    b1 = ldfrag(bbase + (size_t)(kt + 3) * bstride_words, lane);
    if (A_LDS) {
      __builtin_amdgcn_sched_group_barrier(0x100, 2, 0); // 2 DS reads   (A kt+2)
      __builtin_amdgcn_sched_group_barrier(0x020, 2, 0); // 2 VMEM reads (B kt+2)
      __builtin_amdgcn_sched_group_barrier(0x008, 1, 0); // WMMA kt
      __builtin_amdgcn_sched_group_barrier(0x100, 2, 0); // 2 DS reads   (A kt+3)
      __builtin_amdgcn_sched_group_barrier(0x020, 2, 0); // 2 VMEM reads (B kt+3)
      __builtin_amdgcn_sched_group_barrier(0x008, 1, 0); // WMMA kt+1
    } else {
      __builtin_amdgcn_sched_group_barrier(0x020, 4, 0); // 4 VMEM reads (A+B kt+2)
      __builtin_amdgcn_sched_group_barrier(0x008, 1, 0); // WMMA kt
      __builtin_amdgcn_sched_group_barrier(0x020, 4, 0); // 4 VMEM reads (A+B kt+3)
      __builtin_amdgcn_sched_group_barrier(0x008, 1, 0); // WMMA kt+1
    }
  }
  acc = wmma_f(a0, b0, acc);
  acc = wmma_f(a1, b1, acc);
  return acc;
}

// ------------- persistent per-batch-tile RSNN kernel -------------
__global__ __launch_bounds__(1024, 1)
void rsnn_kernel(const uint32_t* __restrict__ xf,
                 const uint32_t* __restrict__ W0f,
                 const uint32_t* __restrict__ W1f,
                 const uint32_t* __restrict__ Wof,
                 const float* __restrict__ tau1,
                 const float* __restrict__ tau2,
                 const float* __restrict__ tau_o,
                 float* __restrict__ out) {
  __shared__ __align__(16) uint32_t bufs[2][CWORDS_];     // 2 x 40 KB ping-pong A chunks
  __shared__ uint32_t bitbuf1[MAXD_ * 256];               // 40 KB bit-packed spikes (layer 1)
  __shared__ uint32_t bitbuf2[MAXD_ * 256];               // 40 KB bit-packed spikes (layer 2)
  __shared__ uint32_t smask1[32 * 8];                     // staged ballot masks (1 KB)
  __shared__ uint32_t smask2[32 * 8];
  __shared__ float o_gemm[16 * 32];
  __shared__ float o_mem_s[16 * NOUT_];
  __shared__ float o_sum_s[16 * NOUT_];
  __shared__ float o_mot_s[16 * NOUT_];
  __shared__ float ao_s[NOUT_];

  const int tid  = threadIdx.x;
  const int lane = tid & 31;
  const int wv   = tid >> 5;           // 32 waves: wave == n-tile of H
  const int g    = blockIdx.x;         // batch tile (16 rows)

  for (int i = tid; i < MAXD_ * 256; i += 1024) { bitbuf1[i] = 0u; bitbuf2[i] = 0u; }
  if (tid < NOUT_) ao_s[tid] = 1.0f / (1.0f + __expf(-tau_o[tid]));
  if (tid < 16 * NOUT_) { o_mem_s[tid] = 0.0f; o_sum_s[tid] = 0.0f; o_mot_s[tid] = 0.0f; }

  const int ncol = wv * 16 + (lane & 15);                 // this lane's hidden column
  const float a1c = 1.0f / (1.0f + __expf(-tau1[ncol]));
  const float a2c = 1.0f / (1.0f + __expf(-tau2[ncol]));

  v8f mem1 = {};
  v8f mem2 = {};
  __syncthreads();

  for (int t = 0; t < WIN_; ++t) {
    const int tslot = t % MAXD_;

    // ---- prev1 chunk 0 expansion (independent of GEMM1; overlaps across waves)
    expand_chunk(bitbuf1, bufs[0], tslot, tid, 0);

    // ---- GEMM1: mem1 = mem1*a1 + x_t @ W0^T (22 k-tiles, global A+B)
    v8f acc1 = gemm_pipe<false>(xf + (size_t)((t * 16 + g) * KT1_) * 256,
                                W0f + (size_t)wv * 256, NT_H_ * 256, KT1_, lane,
                                mem1 * a1c);
    {
      unsigned mask1[8];
      #pragma unroll
      for (int r = 0; r < 8; ++r) {
        float mval = acc1[r];
        bool spk = mval > THRESH_;
        mem1[r] = (mval < THRESH_) ? mval : 0.0f;          // reset; (1-s) decay is implied
        mask1[r] = (unsigned)__ballot(spk);                // rows r (lo16), r+8 (hi16)
      }
      if (lane == 0) {
        #pragma unroll
        for (int r = 0; r < 8; ++r) smask1[wv * 8 + r] = mask1[r];
      }
    }

    // ---- GEMM2 chunks, pipelined against prev1 expansion
    v8f acc2 = mem2 * a2c;
    #pragma unroll 1
    for (int c = 0; c < NCHUNK_; ++c) {
      __syncthreads();                                     // chunk c ready; buf[(c+1)&1] free
      acc2 = gemm_pipe<true>(bufs[c & 1],
                             W1f + ((size_t)(c * CKT_) * NT_H_ + wv) * 256,
                             NT_H_ * 256, CKT_, lane, acc2);
      if (c < NCHUNK_ - 1) expand_chunk(bitbuf1, bufs[(c + 1) & 1], tslot, tid, c + 1);
    }
    {
      unsigned mask2[8];
      #pragma unroll
      for (int r = 0; r < 8; ++r) {
        float mval = acc2[r];
        bool spk = mval > THRESH_;
        mem2[r] = (mval < THRESH_) ? mval : 0.0f;
        mask2[r] = (unsigned)__ballot(spk);
      }
      if (lane == 0) {
        #pragma unroll
        for (int r = 0; r < 8; ++r) smask2[wv * 8 + r] = mask2[r];
      }
    }

    // ---- layer-1 spike write-back: rebuild slot words from staged masks (no atomics)
    __syncthreads();                 // prev1 expansion reads done; smask1 visible
    if (tid < 256) {
      const int m = tid >> 4, w = tid & 15;
      const int r = m & 7;
      const uint32_t lo = smask1[(2 * w    ) * 8 + r];
      const uint32_t hi = smask1[(2 * w + 1) * 8 + r];
      bitbuf1[tslot * 256 + tid] = (m < 8) ? ((lo & 0xFFFFu) | (hi << 16))
                                           : ((lo >> 16) | (hi & 0xFFFF0000u));
    }
    if (tid >= 512) o_gemm[tid - 512] = 0.0f;              // ordered before ds_add by D-loop barrier

    // ---- GEMM3 chunks (K split over waves), pipelined against prev2 expansion
    expand_chunk(bitbuf2, bufs[0], tslot, tid, 0);
    v8f co0 = {};
    v8f co1 = {};
    #pragma unroll 1
    for (int c = 0; c < NCHUNK_; ++c) {
      __syncthreads();
      {
        const uint32_t* bufc = bufs[c & 1];
        int ktl = wv;                          // k-tile within chunk (wave-split)
        int ktg = c * CKT_ + ktl;
        Frag a  = ldfrag(bufc + (size_t)ktl * 256, lane);
        Frag f0 = ldfrag(Wof + (size_t)(ktg * 2 + 0) * 256, lane);
        Frag f1 = ldfrag(Wof + (size_t)(ktg * 2 + 1) * 256, lane);
        co0 = wmma_f(a, f0, co0);
        co1 = wmma_f(a, f1, co1);
        if (wv < CKT_ - 32) {                  // waves 0..7 take a second tile
          ktl = 32 + wv;
          ktg = c * CKT_ + ktl;
          Frag a2  = ldfrag(bufc + (size_t)ktl * 256, lane);
          Frag f02 = ldfrag(Wof + (size_t)(ktg * 2 + 0) * 256, lane);
          Frag f12 = ldfrag(Wof + (size_t)(ktg * 2 + 1) * 256, lane);
          co0 = wmma_f(a2, f02, co0);
          co1 = wmma_f(a2, f12, co1);
        }
      }
      if (c < NCHUNK_ - 1) expand_chunk(bitbuf2, bufs[(c + 1) & 1], tslot, tid, c + 1);
    }
    {
      const int rrow = (lane & 16) ? 8 : 0;
      const int ccol = lane & 15;
      #pragma unroll
      for (int r = 0; r < 8; ++r) {
        atomicAdd(&o_gemm[(r + rrow) * 32 + ccol     ], co0[r]);
        atomicAdd(&o_gemm[(r + rrow) * 32 + 16 + ccol], co1[r]);
      }
    }
    __syncthreads();                 // o_gemm complete; prev2 expansion reads done

    // ---- layer-2 spike write-back (plain stores) + output update, concurrent
    if (tid < 256) {
      const int m = tid >> 4, w = tid & 15;
      const int r = m & 7;
      const uint32_t lo = smask2[(2 * w    ) * 8 + r];
      const uint32_t hi = smask2[(2 * w + 1) * 8 + r];
      bitbuf2[tslot * 256 + tid] = (m < 8) ? ((lo & 0xFFFFu) | (hi << 16))
                                           : ((lo >> 16) | (hi & 0xFFFF0000u));
    }
    if (tid >= 512 && tid < 528) {
      const int row = tid - 512;
      float vals[NOUT_];
      float mx = -3.0e38f;
      #pragma unroll
      for (int o = 0; o < NOUT_; ++o) {
        float nm = o_mem_s[row * NOUT_ + o] * ao_s[o] + o_gemm[row * 32 + o];
        float spk = (nm > THRESH_) ? 1.0f : 0.0f;
        o_sum_s[row * NOUT_ + o] += spk;
        nm = (nm < THRESH_) ? nm : 0.0f;
        o_mem_s[row * NOUT_ + o] = nm;
        vals[o] = nm;
        mx = fmaxf(mx, nm);
      }
      float ssum = 0.0f;
      #pragma unroll
      for (int o = 0; o < NOUT_; ++o) { float e = __expf(vals[o] - mx); vals[o] = e; ssum += e; }
      float inv = 1.0f / ssum;
      #pragma unroll
      for (int o = 0; o < NOUT_; ++o) o_mot_s[row * NOUT_ + o] += vals[o] * inv;
    }
    // next-step chunk-0 barrier orders this against subsequent LDS writers
  }

  __syncthreads();
  if (tid < 16) {
    for (int o = 0; o < NOUT_; ++o) {
      out[(size_t)(g * 16 + tid) * NOUT_ + o] = o_sum_s[tid * NOUT_ + o] * (1.0f / (float)WIN_);
      out[(size_t)B_ * NOUT_ + (size_t)(g * 16 + tid) * NOUT_ + o] = o_mot_s[tid * NOUT_ + o];
    }
  }
}

extern "C" void kernel_launch(void* const* d_in, const int* in_sizes, int n_in,
                              void* d_out, int out_size, void* d_ws, size_t ws_size,
                              hipStream_t stream) {
  (void)in_sizes; (void)n_in; (void)out_size; (void)ws_size;
  const float* input = (const float*)d_in[0];
  const float* W0    = (const float*)d_in[1];
  const float* W1    = (const float*)d_in[2];
  const float* Wo    = (const float*)d_in[3];
  const float* tau1  = (const float*)d_in[4];
  const float* tau2  = (const float*)d_in[5];
  const float* tau_o = (const float*)d_in[6];

  uint32_t* ws  = (uint32_t*)d_ws;
  uint32_t* W0f = ws;
  uint32_t* W1f = W0f + W0F_WORDS;
  uint32_t* Wof = W1f + W1F_WORDS;
  uint32_t* xf  = Wof + WOF_WORDS;

  pack_b_f16<<<(W0F_WORDS + 255) / 256, 256, 0, stream>>>(W0, W0f, H_,    NIN_, KT1_, NT_H_);
  pack_b_f16<<<(W1F_WORDS + 255) / 256, 256, 0, stream>>>(W1, W1f, H_,    HD_,  KT2_, NT_H_);
  pack_b_f16<<<(WOF_WORDS + 255) / 256, 256, 0, stream>>>(Wo, Wof, NOUT_, HD_,  KT2_, NT_O_);
  pack_x_f16<<<(XF_WORDS  + 255) / 256, 256, 0, stream>>>(input, xf);

  rsnn_kernel<<<16, 1024, 0, stream>>>(xf, W0f, W1f, Wof, tau1, tau2, tau_o, (float*)d_out);
}
